// NodeModel_6279242186980
// MI455X (gfx1250) — matrix operand order repaired and checked by reading, not compiled, not run on hardware
//
#include <hip/hip_runtime.h>

#define N_NODES   50000
#define N_EDGES   800000
#define D_NODE_IN 128
#define D_EDGE    96
#define D_GRAPH   64
#define D_IN      288   // 128 + 96 + 64
#define HIDDEN    256
#define D_OUT     128
#define BN_EPS    1e-5f

#define LDS_W1_STRIDE 296   // 288 + 8 pad: 148 dwords == 20 mod 64 -> conflict-free b128 reads
#define LDS_W2_STRIDE 264   // 256 + 8 pad: 132 dwords ==  4 mod 64 -> conflict-free b128 reads

typedef __bf16 v16bf __attribute__((ext_vector_type(16)));
typedef float  v8f   __attribute__((ext_vector_type(8)));
typedef int    v4i   __attribute__((ext_vector_type(4)));

union V16U { uint4 q[2]; unsigned u[8]; v16bf v; };

__device__ __forceinline__ unsigned short f2bf(float f) {
    unsigned u = __builtin_bit_cast(unsigned, f);
    u += 0x7fffu + ((u >> 16) & 1u);           // round-to-nearest-even
    return (unsigned short)(u >> 16);
}
__device__ __forceinline__ float bf2f(unsigned v) {
    return __builtin_bit_cast(float, v << 16);
}
__device__ __forceinline__ v16bf mkv(uint4 a, uint4 b) {
    V16U t; t.q[0] = a; t.q[1] = b; return t.v;
}

// 16-byte global->LDS copy: async (ASYNCcnt) path if the toolchain exposes it
__device__ __forceinline__ void async_copy16(const unsigned short* g, unsigned short* l) {
#if __has_builtin(__builtin_amdgcn_global_load_async_to_lds_b128)
    __builtin_amdgcn_global_load_async_to_lds_b128(
        (__attribute__((address_space(1))) v4i*)g,
        (__attribute__((address_space(3))) v4i*)l, 0, 0);
#else
    *(uint4*)l = *(const uint4*)g;             // sync fallback: global_load + ds_store
#endif
}
__device__ __forceinline__ void wait_async_copies() {
#if __has_builtin(__builtin_amdgcn_global_load_async_to_lds_b128)
#if __has_builtin(__builtin_amdgcn_s_wait_asynccnt)
    __builtin_amdgcn_s_wait_asynccnt(0);
#else
    asm volatile("s_wait_asynccnt 0" ::: "memory");
#endif
#endif
}

// ---------------------------------------------------------------- utilities
__global__ void zero_kernel(float* __restrict__ p, long long n) {
    long long i = (long long)blockIdx.x * blockDim.x + threadIdx.x;
    if (i < n) p[i] = 0.0f;
}

// ------------------------------------------------------ 1) edge scatter-sum
// one thread per (edge, 4-dim chunk): float4 load + 4 L2-resident f32 atomics
__global__ __launch_bounds__(256) void edge_scatter(
    const float* __restrict__ edge_attr, const int* __restrict__ edge_index,
    float* __restrict__ sums, float* __restrict__ cnt)
{
    long long i = (long long)blockIdx.x * 256 + threadIdx.x;
    if (i >= (long long)N_EDGES * (D_EDGE / 4)) return;
    int e = (int)(i / (D_EDGE / 4));
    int c = (int)(i - (long long)e * (D_EDGE / 4));
    int col = edge_index[N_EDGES + e];          // edge_index[1][e]
    float4 v = *(const float4*)(edge_attr + (long long)e * D_EDGE + c * 4);
    float* dst = sums + (long long)col * D_EDGE + c * 4;
    atomicAdd(dst + 0, v.x); atomicAdd(dst + 1, v.y);
    atomicAdd(dst + 2, v.z); atomicAdd(dst + 3, v.w);
    if (c == 0) atomicAdd(&cnt[col], 1.0f);
}

// ------------------------------------- 2) Hin = concat(x, mean, u[batch]) bf16
__global__ __launch_bounds__(256) void build_hin(
    const float* __restrict__ x, const float* __restrict__ sums,
    const float* __restrict__ cnt, const float* __restrict__ u,
    const int* __restrict__ batch, unsigned short* __restrict__ Hin)
{
    long long i = (long long)blockIdx.x * 256 + threadIdx.x;
    if (i >= (long long)N_NODES * D_IN) return;
    int n = (int)(i / D_IN);
    int d = (int)(i - (long long)n * D_IN);
    float v;
    if (d < D_NODE_IN) {
        v = x[(long long)n * D_NODE_IN + d];
    } else if (d < D_NODE_IN + D_EDGE) {
        float c = fmaxf(cnt[n], 1.0f);
        v = sums[(long long)n * D_EDGE + (d - D_NODE_IN)] * (1.0f / c);
    } else {
        v = u[batch[n] * D_GRAPH + (d - D_NODE_IN - D_EDGE)];
    }
    Hin[i] = f2bf(v);
}

// ------------------------------- 3) transpose + convert weights to bf16 (N,K)
__global__ __launch_bounds__(256) void prep_weights(
    const float* __restrict__ W1, const float* __restrict__ W2,
    unsigned short* __restrict__ W1T, unsigned short* __restrict__ W2T)
{
    int i = blockIdx.x * 256 + threadIdx.x;
    if (i < HIDDEN * D_IN) {
        int n = i / D_IN, k = i - n * D_IN;
        W1T[i] = f2bf(W1[k * HIDDEN + n]);      // W1 is [288,256] row-major
    } else {
        int j = i - HIDDEN * D_IN;
        if (j < D_OUT * HIDDEN) {
            int n = j / HIDDEN, k = j - n * HIDDEN;
            W2T[j] = f2bf(W2[k * D_OUT + n]);   // W2 is [256,128] row-major
        }
    }
}

// ------- 4) GEMM1 (bf16 WMMA, f32 acc) + BN-stat reduction, LDS-staged weights
// block = 256 threads = 8 waves arranged 2(M) x 4(N): 32 rows x 256 cols / block.
// W1T (147KB) async-copied once per block into padded LDS.
__global__ __launch_bounds__(256) void gemm1_kernel(
    const unsigned short* __restrict__ Hin, const unsigned short* __restrict__ W1T,
    unsigned short* __restrict__ h1, float* __restrict__ bn_sum, float* __restrict__ bn_sq)
{
    extern __shared__ char smem_raw[];
    unsigned short* ldsW = (unsigned short*)smem_raw;      // [256][LDS_W1_STRIDE]

    // cooperative async fill: 256 rows x 36 chunks of 8 bf16
    for (int c = threadIdx.x; c < 256 * 36; c += 256) {
        int row = c / 36;
        int off = (c - row * 36) * 8;
        async_copy16(W1T + row * D_IN + off, ldsW + row * LDS_W1_STRIDE + off);
    }
    wait_async_copies();
    __syncthreads();

    const int lane   = threadIdx.x & 31;
    const int wave   = threadIdx.x >> 5;
    const int n_wave = (wave & 3) * 64;
    const int m_base = blockIdx.x * 32 + (wave >> 2) * 16;
    const int ln     = lane & 15;
    const bool hiH   = lane >= 16;
    if (m_base >= N_NODES) return;                         // tail block: wave-uniform skip

    v8f acc[4] = {v8f{}, v8f{}, v8f{}, v8f{}};
    // A layout (16-bit A 16x32): lane holds M=ln; elems 0-7 -> K = s*32+hi+0..7,
    // elems 8-15 -> K = s*32+16+hi+0..7  (hi = 8 for lanes 16-31)
    const unsigned short* arow = Hin + (long long)(m_base + ln) * D_IN + (hiH ? 8 : 0);

    #pragma unroll
    for (int s = 0; s < D_IN / 32; ++s) {
        const uint4* ap = (const uint4*)(arow + s * 32);
        v16bf A = mkv(ap[0], ap[2]);                       // +16 elems == +2 uint4
        #pragma unroll
        for (int t = 0; t < 4; ++t) {
            // B layout (16-bit B 32x16): lane holds N=ln; elems 0-7 -> K = s*32+hi+0..7, elems 8-15 -> K = s*32+16+hi+0..7
            const uint4* bp = (const uint4*)(ldsW + (n_wave + t * 16 + ln) * LDS_W1_STRIDE
                                             + s * 32 + (hiH ? 8 : 0));
            v16bf B = mkv(bp[0], bp[2]);
            acc[t] = __builtin_amdgcn_wmma_f32_16x16x32_bf16(
                false, A, false, B, (short)0, acc[t], false, false);
        }
    }

    // Epilogue: C layout -> lane L, elem r: M = m_base + r + (hiH?8:0), N = col
    #pragma unroll
    for (int t = 0; t < 4; ++t) {
        int n = n_wave + t * 16 + ln;
        int mh = hiH ? 8 : 0;
        float s = 0.0f, q = 0.0f;
        #pragma unroll
        for (int r = 0; r < 8; ++r) {
            float v = acc[t][r];
            s += v; q += v * v;
            h1[(long long)(m_base + r + mh) * HIDDEN + n] = f2bf(v);
        }
        s += __shfl_xor(s, 16, 32);             // combine the two M-halves per column
        q += __shfl_xor(q, 16, 32);
        if (!hiH) {
            atomicAdd(&bn_sum[n], s);
            atomicAdd(&bn_sq[n], q);
        }
    }
}

// ---------------------------------------------- 5) BN finalize -> scale/shift
__global__ void bn_finalize(
    const float* __restrict__ bn_sum, const float* __restrict__ bn_sq,
    const float* __restrict__ gamma, const float* __restrict__ beta,
    float* __restrict__ scale, float* __restrict__ shift)
{
    int c = threadIdx.x;
    const float invN = 1.0f / (float)N_NODES;
    float mu  = bn_sum[c] * invN;
    float var = bn_sq[c] * invN - mu * mu;
    float inv = rsqrtf(var + BN_EPS);
    float sc  = gamma[c] * inv;
    scale[c] = sc;
    shift[c] = beta[c] - mu * sc;
}

// --- 6) GEMM2: fused BN+ReLU on A while repacking bf16; LDS-staged W2T; +b2
// block = 128 threads = 4 waves arranged 2(M) x 2(N): 32 rows x 128 cols / block.
__global__ __launch_bounds__(128) void gemm2_kernel(
    const unsigned short* __restrict__ h1, const unsigned short* __restrict__ W2T,
    const float* __restrict__ scale, const float* __restrict__ shift,
    const float* __restrict__ b2, float* __restrict__ out)
{
    extern __shared__ char smem_raw[];
    unsigned short* ldsW = (unsigned short*)smem_raw;      // [128][LDS_W2_STRIDE]

    for (int c = threadIdx.x; c < 128 * 32; c += 128) {    // 128 rows x 32 chunks of 8
        int row = c >> 5;
        int off = (c & 31) * 8;
        async_copy16(W2T + row * HIDDEN + off, ldsW + row * LDS_W2_STRIDE + off);
    }
    wait_async_copies();
    __syncthreads();

    const int lane   = threadIdx.x & 31;
    const int wave   = threadIdx.x >> 5;
    const int n_wave = (wave & 1) * 64;
    const int m_base = blockIdx.x * 32 + (wave >> 1) * 16;
    const int ln     = lane & 15;
    const bool hiH   = lane >= 16;
    const int hi     = hiH ? 8 : 0;
    if (m_base >= N_NODES) return;                         // tail block: wave-uniform skip

    v8f acc[4] = {v8f{}, v8f{}, v8f{}, v8f{}};
    const unsigned short* arow = h1 + (long long)(m_base + ln) * HIDDEN;

    #pragma unroll
    for (int s = 0; s < HIDDEN / 32; ++s) {
        int k0 = s * 32;
        V16U au;
        #pragma unroll
        for (int g = 0; g < 2; ++g) {           // elems g*8 .. g*8+7 : K = k0 + g*16 + hi + 0..7
            int kb = k0 + g * 16 + hi;
            uint4 raw = *(const uint4*)(arow + kb);
            float4 sc0 = *(const float4*)(scale + kb);
            float4 sc1 = *(const float4*)(scale + kb + 4);
            float4 sh0 = *(const float4*)(shift + kb);
            float4 sh1 = *(const float4*)(shift + kb + 4);
            unsigned rr[4] = {raw.x, raw.y, raw.z, raw.w};
            float scv[8] = {sc0.x, sc0.y, sc0.z, sc0.w, sc1.x, sc1.y, sc1.z, sc1.w};
            float shv[8] = {sh0.x, sh0.y, sh0.z, sh0.w, sh1.x, sh1.y, sh1.z, sh1.w};
            #pragma unroll
            for (int p = 0; p < 4; ++p) {
                float e0 = fmaxf(fmaf(bf2f(rr[p] & 0xffffu), scv[2 * p],     shv[2 * p]),     0.0f);
                float e1 = fmaxf(fmaf(bf2f(rr[p] >> 16),     scv[2 * p + 1], shv[2 * p + 1]), 0.0f);
                au.u[g * 4 + p] = (unsigned)f2bf(e0) | ((unsigned)f2bf(e1) << 16);
            }
        }
        v16bf A = au.v;
        #pragma unroll
        for (int t = 0; t < 4; ++t) {
            const uint4* bp = (const uint4*)(ldsW + (n_wave + t * 16 + ln) * LDS_W2_STRIDE
                                             + k0 + (hiH ? 8 : 0));
            v16bf B = mkv(bp[0], bp[2]);
            acc[t] = __builtin_amdgcn_wmma_f32_16x16x32_bf16(
                false, A, false, B, (short)0, acc[t], false, false);
        }
    }

    #pragma unroll
    for (int t = 0; t < 4; ++t) {
        int n = n_wave + t * 16 + ln;
        float bb = b2[n];
        int mh = hiH ? 8 : 0;
        #pragma unroll
        for (int r = 0; r < 8; ++r)
            out[(long long)(m_base + r + mh) * D_OUT + n] = acc[t][r] + bb;
    }
}

// --------------------------------------------------------------------------
extern "C" void kernel_launch(void* const* d_in, const int* in_sizes, int n_in,
                              void* d_out, int out_size, void* d_ws, size_t ws_size,
                              hipStream_t stream) {
    const float* x          = (const float*)d_in[0];
    const int*   edge_index = (const int*)  d_in[1];
    const float* edge_attr  = (const float*)d_in[2];
    const float* u          = (const float*)d_in[3];
    const int*   batch      = (const int*)  d_in[4];
    const float* W1         = (const float*)d_in[5];
    // d_in[6] = b1: cancels exactly under training-mode BatchNorm -> unused
    const float* gamma      = (const float*)d_in[7];
    const float* beta       = (const float*)d_in[8];
    const float* W2         = (const float*)d_in[9];
    const float* b2         = (const float*)d_in[10];
    float* out = (float*)d_out;

    // Workspace layout (region [0, 25.6MB) is reused: sums/cnt first, then h1 bf16)
    char* ws = (char*)d_ws;
    float*          sums  = (float*)(ws + 0);                      // 50000*96*4  = 19,200,000
    float*          cnt   = (float*)(ws + 19200000);               // 50000*4     =    200,000
    unsigned short* h1bf  = (unsigned short*)(ws + 0);             // 50000*256*2 = 25,600,000
    unsigned short* Hin   = (unsigned short*)(ws + 25600000);      // 50000*288*2 = 28,800,000
    unsigned short* W1T   = (unsigned short*)(ws + 54400000);      // 256*288*2   =    147,456
    unsigned short* W2T   = (unsigned short*)(ws + 54547456);      // 128*256*2   =     65,536
    float*          bn_s  = (float*)(ws + 54612992);               // 256
    float*          bn_q  = bn_s + 256;
    float*          scale = bn_s + 512;
    float*          shift = bn_s + 768;

    // 0) zero accumulators (sums+cnt are contiguous)
    {
        long long n = 4850000;  // 4.8M sums + 50K cnt
        zero_kernel<<<(unsigned)((n + 255) / 256), 256, 0, stream>>>(sums, n);
        zero_kernel<<<2, 256, 0, stream>>>(bn_s, 512);
    }
    // 1) scatter-sum edges (L2-resident atomics)
    edge_scatter<<<(N_EDGES * (D_EDGE / 4)) / 256, 256, 0, stream>>>(edge_attr, edge_index, sums, cnt);
    // 2) build concat input in bf16
    build_hin<<<(N_NODES * D_IN) / 256, 256, 0, stream>>>(x, sums, cnt, u, batch, Hin);
    // 3) weight transpose/convert
    prep_weights<<<(HIDDEN * D_IN + D_OUT * HIDDEN) / 256, 256, 0, stream>>>(W1, W2, W1T, W2T);
    // 4) GEMM1 + BN statistics (32 rows/block, LDS-staged W1T)
    {
        unsigned grid = (N_NODES + 31) / 32;   // 1563
        size_t   lds  = 256 * LDS_W1_STRIDE * sizeof(unsigned short);  // 151,552 B
        gemm1_kernel<<<grid, 256, lds, stream>>>(Hin, W1T, h1bf, bn_s, bn_q);
    }
    // 5) BN scale/shift
    bn_finalize<<<1, HIDDEN, 0, stream>>>(bn_s, bn_q, gamma, beta, scale, shift);
    // 6) GEMM2 with fused BN+ReLU (32 rows/block, LDS-staged W2T)
    {
        unsigned grid = (N_NODES + 31) / 32;   // 1563
        size_t   lds  = 128 * LDS_W2_STRIDE * sizeof(unsigned short);  // 67,584 B
        gemm2_kernel<<<grid, 128, lds, stream>>>(h1bf, W2T, scale, shift, b2, out);
    }
}
